// UnidirectionalAttention_49821620634024
// MI455X (gfx1250) — matrix operand order repaired and checked by reading, not compiled, not run on hardware
//
#include <hip/hip_runtime.h>
#include <hip/hip_bf16.h>

typedef __bf16 bf16;
typedef __attribute__((ext_vector_type(16))) __bf16 v16bf;
typedef __attribute__((ext_vector_type(8)))  __bf16 v8bf;
typedef __attribute__((ext_vector_type(8)))  float  v8f;
typedef __attribute__((ext_vector_type(4)))  float  v4f;
typedef __attribute__((ext_vector_type(4)))  unsigned int u32x4;
typedef __attribute__((ext_vector_type(8)))  int i32x8;
typedef __attribute__((ext_vector_type(4)))  int i32x4;

#define HIDDEN 1024
#define NHEADS 16
#define HEADD  64
#define SEQ    2048
#define BATCH  2

union bfpack { unsigned u; bf16 h[2]; };

__device__ __forceinline__ v8f wmma_bf16(v16bf a, v16bf b, v8f c) {
  return __builtin_amdgcn_wmma_f32_16x16x32_bf16(false, a, false, b, (short)0, c,
                                                 false, false);
}

// B-fragment from bf16 LDS stored [col][k]: bf16 p[0..7] and p[16..23].
__device__ __forceinline__ v16bf ldsB_frag(const bf16* p) {
  v8bf lo = *(const v8bf*)(p);
  v8bf hi = *(const v8bf*)(p + 16);
  return __builtin_shufflevector(lo, hi, 0,1,2,3,4,5,6,7,8,9,10,11,12,13,14,15);
}

// A-fragment from f32 LDS row (TDM-staged): elements 0..7 <- row[kb..kb+7],
// 8..15 <- row[kb+16..kb+23], converted to bf16.
__device__ __forceinline__ v16bf ldsA_frag_f32(const float* row, int kb) {
  v4f a = *(const v4f*)(row + kb);
  v4f b = *(const v4f*)(row + kb + 4);
  v4f c = *(const v4f*)(row + kb + 16);
  v4f d = *(const v4f*)(row + kb + 20);
  v16bf r;
#pragma unroll
  for (int e = 0; e < 4; ++e) {
    r[e]      = (bf16)a[e];
    r[e + 4]  = (bf16)b[e];
    r[e + 8]  = (bf16)c[e];
    r[e + 12] = (bf16)d[e];
  }
  return r;
}

// B-fragment from f32 LDS (TDM-staged K tile): f32 p[0..7], p[16..23] -> bf16.
__device__ __forceinline__ v16bf ldsB_frag_f32(const float* p) {
  v4f a = *(const v4f*)(p);
  v4f b = *(const v4f*)(p + 4);
  v4f c = *(const v4f*)(p + 16);
  v4f d = *(const v4f*)(p + 20);
  v16bf r;
#pragma unroll
  for (int e = 0; e < 4; ++e) {
    r[e]      = (bf16)a[e];
    r[e + 4]  = (bf16)b[e];
    r[e + 8]  = (bf16)c[e];
    r[e + 12] = (bf16)d[e];
  }
  return r;
}

// A-fragment from bf16 LDS (P scratch), row-major stride in elements.
__device__ __forceinline__ v16bf ldsA_frag(const bf16* row, int kb) {
  v8bf lo = *(const v8bf*)(row + kb);
  v8bf hi = *(const v8bf*)(row + kb + 16);
  return __builtin_shufflevector(lo, hi, 0,1,2,3,4,5,6,7,8,9,10,11,12,13,14,15);
}

// ---------------------------------------------------------------------------
// TDM: async 2D tile load global->LDS (f32 elements, optional LDS padding).
// Descriptor per CDNA5 ISA (D# group0 + group1).
// This toolchain exposes the 6-arg builtin:
//   (u32x4 g0, i32x8 g1, i32x4 g2, i32x4 g3, i32x8 extra, i32 cpol)
// groups 2/3 (and the extra group) are unused for 2D tiles -> zeros.
// ---------------------------------------------------------------------------
__device__ __forceinline__ void tdm_load_2d_f32(
    const float* gaddr, unsigned lds_off,
    unsigned tile_d0, unsigned tile_d1, unsigned stride0_elems,
    unsigned pad_interval_enum, unsigned pad_amount_m1) {
  unsigned long long ga = (unsigned long long)(const void*)gaddr;
  u32x4 g0;
  g0[0] = 1u;                                            // count=1, user desc
  g0[1] = lds_off;                                       // lds_addr (bytes)
  g0[2] = (unsigned)ga;                                  // global_addr[31:0]
  g0[3] = (unsigned)((ga >> 32) & 0x1FFFFFFu) | (2u << 30); // addr[56:32], type=2
  i32x8 g1;
  g1[0] = (int)((2u << 16) |                             // data_size = 4B
                (1u << 20) |                             // pad_enable
                (pad_interval_enum << 22) |
                (pad_amount_m1 << 25));
  g1[1] = (int)(tile_d0 << 16);                          // tensor_dim0 = tile_d0
  g1[2] = (int)((tile_d0 >> 16) | (tile_d1 << 16));      // tensor_dim1 = tile_d1
  g1[3] = (int)((tile_d1 >> 16) | (tile_d0 << 16));      // tile_dim0
  g1[4] = (int)(tile_d1 & 0xFFFFu);                      // tile_dim1 (tile_dim2=0)
  g1[5] = (int)stride0_elems;                            // tensor_dim0_stride lo
  g1[6] = 0;                                             // stride hi / dim1_stride
  g1[7] = 0;
  i32x4 z4 = {0, 0, 0, 0};
  i32x8 z8 = {0, 0, 0, 0, 0, 0, 0, 0};
  __builtin_amdgcn_tensor_load_to_lds(g0, g1, z4, z4, z8, 0);
}

// ---------------------------------------------------------------------------
// C[M,N] = A[M,K] @ B[K,N] + bias[N]   (f32 in/out, bf16 WMMA compute)
// block 256 thr = 8 waves; block tile 64x256; wave tile 32x64 (2x4 WMMA).
// A tile staged f32 by TDM (row-major, padded 32->36 dwords);
// B tile transpose-converted to bf16 by threads (packed b32 stores).
// ---------------------------------------------------------------------------
__global__ __launch_bounds__(256)
void gemm_bias_wmma(const float* __restrict__ A, const float* __restrict__ B,
                    const float* __restrict__ bias, float* __restrict__ C,
                    int M, int N, int K) {
  __shared__ __align__(16) float As[64 * 36];   // TDM-staged [m][k], stride 36
  __shared__ __align__(16) bf16  Bs[256 * 40];  // [n][k] transposed, stride 40

  const int tid = threadIdx.x;
  const int wave = tid >> 5;
  const int lane = tid & 31;
  const int lane15 = lane & 15;
  const int m0 = blockIdx.y * 64;
  const int n0 = blockIdx.x * 256;
  const int wr = (wave >> 2) * 32;
  const int wc = (wave & 3) * 64;
  const int akb = (lane < 16) ? 0 : 8;
  const int bkb = (lane < 16) ? 0 : 8;
  const int cmb = (lane < 16) ? 0 : 8;

  v8f acc[2][4] = {};

  const int kp = (tid >> 4) * 2;   // B stage: k pair 0..30
  const int nc = (tid & 15) * 16;  // B stage: 16 n per thread

  const unsigned as_off = (unsigned)(size_t)(void*)As;

  for (int k0 = 0; k0 < K; k0 += 32) {
    // async DMA of the 64x32 A tile (one issue per workgroup)
    if (wave == 0) {
      tdm_load_2d_f32(A + (size_t)m0 * K + k0, as_off,
                      /*tile_d0=*/32, /*tile_d1=*/64, /*stride0=*/(unsigned)K,
                      /*pad_interval(32dw)=*/4, /*pad_amount(4dw)=*/3);
    }
    // overlap: threads transpose-convert the 32x256 B tile
    {
      const float* b0 = B + (size_t)(k0 + kp) * N + n0 + nc;
      const float* b1 = b0 + N;
#pragma unroll
      for (int i = 0; i < 16; ++i) {
        bfpack pk;
        pk.h[0] = (bf16)b0[i];
        pk.h[1] = (bf16)b1[i];
        *(unsigned*)&Bs[(nc + i) * 40 + kp] = pk.u;
      }
      if (k0 + 32 < K) __builtin_prefetch(b0 + (size_t)32 * N, 0, 0);
    }
    if (wave == 0) __builtin_amdgcn_s_wait_tensorcnt(0);
    __syncthreads();

    v16bf af[2], bfr[4];
#pragma unroll
    for (int i = 0; i < 2; ++i)
      af[i] = ldsA_frag_f32(As + (wr + i * 16 + lane15) * 36, akb);
#pragma unroll
    for (int j = 0; j < 4; ++j)
      bfr[j] = ldsB_frag(Bs + (wc + j * 16 + lane15) * 40 + bkb);

#pragma unroll
    for (int i = 0; i < 2; ++i)
#pragma unroll
      for (int j = 0; j < 4; ++j)
        acc[i][j] = wmma_bf16(af[i], bfr[j], acc[i][j]);
    __syncthreads();
  }

#pragma unroll
  for (int i = 0; i < 2; ++i)
#pragma unroll
    for (int j = 0; j < 4; ++j) {
      const int col = n0 + wc + j * 16 + lane15;
      const float bv = bias[col];
#pragma unroll
      for (int r = 0; r < 8; ++r) {
        const int row = m0 + wr + i * 16 + cmb + r;
        C[(size_t)row * N + col] = acc[i][j][r] + bv;
      }
    }
}

// ---------------------------------------------------------------------------
// Causal flash attention over qkv[B,S,3H]; writes attn_out[B,S,H] (f32).
// grid (S/64, B*NHEADS), block 128 (4 waves); wave owns 16 queries.
// K tile (32x64, row-major) staged f32 by TDM (padded 64->68 dwords);
// V tile transpose-converted to bf16 by threads.
// ---------------------------------------------------------------------------
__global__ __launch_bounds__(128)
void attn_wmma(const float* __restrict__ qkv, float* __restrict__ out) {
  __shared__ __align__(16) float Ksf[32 * 68];     // TDM-staged [key][hd]
  __shared__ __align__(16) bf16  Vt[64 * 40];      // [hd][key], stride 40
  __shared__ __align__(16) bf16  Ps[4 * 16 * 40];  // per-wave P scratch

  const int tid = threadIdx.x;
  const int wave = tid >> 5;
  const int lane = tid & 31;
  const int lane15 = lane & 15;
  const int akb = (lane < 16) ? 0 : 8;
  const int bkb = (lane < 16) ? 0 : 8;
  const int cmb = (lane < 16) ? 0 : 8;

  const int b = blockIdx.y >> 4;   // NHEADS = 16
  const int h = blockIdx.y & 15;
  const int qb = blockIdx.x * 64;
  const int q0 = qb + wave * 16;
  const int q_hi = q0 + 15;

  const size_t rs = 3 * HIDDEN;
  const float scale = 0.125f;      // 1/sqrt(64), folded into Q

  const float* qrow = qkv + (size_t)(b * SEQ + q0 + lane15) * rs + h * HEADD;
  v16bf qf[2];
#pragma unroll
  for (int f = 0; f < 2; ++f)
#pragma unroll
    for (int e = 0; e < 8; ++e) {
      qf[f][e]     = (bf16)(qrow[f * 32 + akb + e] * scale);
      qf[f][e + 8] = (bf16)(qrow[f * 32 + akb + 16 + e] * scale);
    }

  v8f acc[4] = {};
  float m_i[8], l_i[8];
#pragma unroll
  for (int r = 0; r < 8; ++r) { m_i[r] = -1e30f; l_i[r] = 0.0f; }

  const int vkp = (tid & 15) * 2;   // V stage: key pair
  const int vhc = (tid >> 4) * 8;   // V stage: 8 hd per thread
  bf16* Pw = Ps + wave * 16 * 40;
  const unsigned ks_off = (unsigned)(size_t)(void*)Ksf;

  for (int j0 = 0; j0 <= qb + 63; j0 += 32) {
    if (wave == 0) {
      tdm_load_2d_f32(qkv + (size_t)(b * SEQ + j0) * rs + HIDDEN + h * HEADD,
                      ks_off, /*tile_d0=*/64, /*tile_d1=*/32,
                      /*stride0=*/(unsigned)rs,
                      /*pad_interval(64dw)=*/5, /*pad_amount(4dw)=*/3);
    }
    {
      const float* v0 = qkv + (size_t)(b * SEQ + j0 + vkp) * rs + 2 * HIDDEN + h * HEADD + vhc;
      const float* v1 = v0 + rs;
#pragma unroll
      for (int i = 0; i < 8; ++i) {
        bfpack pk;
        pk.h[0] = (bf16)v0[i];
        pk.h[1] = (bf16)v1[i];
        *(unsigned*)&Vt[(vhc + i) * 40 + vkp] = pk.u;
      }
    }
    if (wave == 0) __builtin_amdgcn_s_wait_tensorcnt(0);
    __syncthreads();

    if (j0 <= q_hi) {
      v16bf k0lo = ldsB_frag_f32(Ksf + lane15 * 68 + bkb);
      v16bf k0hi = ldsB_frag_f32(Ksf + lane15 * 68 + 32 + bkb);
      v16bf k1lo = ldsB_frag_f32(Ksf + (16 + lane15) * 68 + bkb);
      v16bf k1hi = ldsB_frag_f32(Ksf + (16 + lane15) * 68 + 32 + bkb);
      v8f sa = {}, sb = {};
      sa = wmma_bf16(qf[0], k0lo, sa);
      sa = wmma_bf16(qf[1], k0hi, sa);
      sb = wmma_bf16(qf[0], k1lo, sb);
      sb = wmma_bf16(qf[1], k1hi, sb);

#pragma unroll
      for (int r = 0; r < 8; ++r) {
        const int q = q0 + cmb + r;
        if (j0 + lane15 > q)      sa[r] = -10000.0f;
        if (j0 + 16 + lane15 > q) sb[r] = -10000.0f;
      }

#pragma unroll
      for (int r = 0; r < 8; ++r) {
        float mx = fmaxf(sa[r], sb[r]);
#pragma unroll
        for (int d = 1; d < 16; d <<= 1)
          mx = fmaxf(mx, __shfl_xor(mx, d, 32));
        const float mnew = fmaxf(m_i[r], mx);
        const float corr = __expf(m_i[r] - mnew);
        const float pa = __expf(sa[r] - mnew);
        const float pb = __expf(sb[r] - mnew);
        float ps = pa + pb;
#pragma unroll
        for (int d = 1; d < 16; d <<= 1)
          ps += __shfl_xor(ps, d, 32);
        l_i[r] = l_i[r] * corr + ps;
        m_i[r] = mnew;
#pragma unroll
        for (int t = 0; t < 4; ++t) acc[t][r] *= corr;
        Pw[(cmb + r) * 40 + lane15]      = (bf16)pa;
        Pw[(cmb + r) * 40 + 16 + lane15] = (bf16)pb;
      }
      asm volatile("" ::: "memory");   // DS stores precede DS loads (in-order per wave)

      v16bf pf = ldsA_frag(Pw + lane15 * 40, akb);
#pragma unroll
      for (int t = 0; t < 4; ++t) {
        v16bf vf = ldsB_frag(Vt + (t * 16 + lane15) * 40 + bkb);
        acc[t] = wmma_bf16(pf, vf, acc[t]);
      }
    }
    __syncthreads();
  }

#pragma unroll
  for (int r = 0; r < 8; ++r) {
    const float inv = 1.0f / l_i[r];
    const int q = q0 + cmb + r;
    float* orow = out + (size_t)(b * SEQ + q) * HIDDEN + h * HEADD;
#pragma unroll
    for (int t = 0; t < 4; ++t)
      orow[t * 16 + lane15] = acc[t][r] * inv;
  }
}

extern "C" void kernel_launch(void* const* d_in, const int* in_sizes, int n_in,
                              void* d_out, int out_size, void* d_ws, size_t ws_size,
                              hipStream_t stream) {
  const float* x      = (const float*)d_in[0];
  const float* w_attn = (const float*)d_in[1];
  const float* b_attn = (const float*)d_in[2];
  const float* w_proj = (const float*)d_in[3];
  const float* b_proj = (const float*)d_in[4];
  float* out = (float*)d_out;

  float* qkv  = (float*)d_ws;                              // [B*S, 3H]
  float* attn = qkv + (size_t)BATCH * SEQ * 3 * HIDDEN;    // [B*S, H]

  const int M = BATCH * SEQ;   // 4096

  gemm_bias_wmma<<<dim3(3 * HIDDEN / 256, M / 64), dim3(256), 0, stream>>>(
      x, w_attn, b_attn, qkv, M, 3 * HIDDEN, HIDDEN);

  attn_wmma<<<dim3(SEQ / 64, BATCH * NHEADS), dim3(128), 0, stream>>>(qkv, attn);

  gemm_bias_wmma<<<dim3(HIDDEN / 256, M / 64), dim3(256), 0, stream>>>(
      attn, w_proj, b_proj, out, M, HIDDEN, HIDDEN);
}
